// FALayer_8710193676510
// MI455X (gfx1250) — compile-verified, matrix operand order checked by reading, not run on hardware
//
#include <hip/hip_runtime.h>
#include <cstdint>

// ---------------------------------------------------------------------------
// FALayer forward for MI455X (gfx1250, wave32).
// Hot loop: [E,48] x [48,48] GEMM via v_wmma_f32_16x16x32_bf16
// (M=16 edges/wave, N=48 -> 3 tiles, K=48 zero-padded to 64 -> 2 k-steps).
// W1 B-fragments + b1/W2 scalars live in registers (loop-invariant).
// Epilogue reduction is an LDS transpose (8 ds_store_b32 + 4 ds_load_b128,
// one wait) instead of 32 serialized ds_bpermute chains; tanh runs on 16
// lanes only. Scatter = HW global_atomic_add_f32 at agent scope (L2).
// h (19.2MB) and z (19.2MB) are L2-resident on the 192MB L2.
// ---------------------------------------------------------------------------

typedef __attribute__((ext_vector_type(16))) __bf16 v16bf;
typedef __attribute__((ext_vector_type(2)))  __bf16 bf16x2;
typedef __attribute__((ext_vector_type(8)))  float  v8f;

#define DD     48
#define KPAD   64
#define ROWW   72      // 64 bf16 + 8 pad shorts (144B rows, 16B aligned)
#define WAVES  8
#define TILE_M 16

union FragAB {
    uint4 q[2];       // two ds_load_b128
    v16bf v;          // 16 bf16 = 8 VGPRs
};

// native packed bf16 convert (lets backend emit v_cvt_pk_bf16_f32)
__device__ __forceinline__ unsigned int pack_bf16(float a, float b) {
    bf16x2 v;
    v.x = (__bf16)a;
    v.y = (__bf16)b;
    return __builtin_bit_cast(unsigned int, v);
}

__device__ __forceinline__ unsigned short f2bf(float f) {
    __bf16 x = (__bf16)f;
    return __builtin_bit_cast(unsigned short, x);
}

__global__ void zero_kernel(float* __restrict__ p, long n) {
    long i      = (long)blockIdx.x * blockDim.x + threadIdx.x;
    long stride = (long)gridDim.x * blockDim.x;
    for (; i < n; i += stride) p[i] = 0.0f;
}

__global__ __launch_bounds__(256) void falayer_kernel(
    const float* __restrict__ h,   const int* __restrict__ src,
    const int*   __restrict__ dst, const float* __restrict__ W1,
    const float* __restrict__ b1,  const float* __restrict__ alphaP,
    const float* __restrict__ W2,  const float* __restrict__ b2P,
    float* __restrict__ z, float* __restrict__ e_out,
    long E, int nTiles)
{
    __shared__ __align__(16) unsigned short sW1[DD][ROWW];
    __shared__ __align__(16) unsigned short sH2[WAVES][TILE_M][ROWW];
    __shared__ __align__(16) float sP[WAVES][2][8][16]; // [wv][hi][r][nl]
    __shared__ float sG[WAVES][TILE_M];

    const int tid = threadIdx.x;

    // Stage W1 -> bf16 LDS once per block ([n][k], zero-pad k = 48..63)
    for (int idx = tid; idx < DD * KPAD; idx += 256) {
        int n = idx >> 6, k = idx & 63;
        sW1[n][k] = (k < DD) ? f2bf(W1[n * DD + k]) : (unsigned short)0;
    }
    __syncthreads();

    const float alpha = alphaP[0];
    const float b2    = b2P[0];

    const int wv     = tid >> 5;
    const int lane   = tid & 31;
    const int nl     = lane & 15;   // column within WMMA tile
    const int hi     = lane >> 4;   // half-wave select
    const int kb     = hi * 8;      // K-pair base for this half-wave
    const int m_pair = lane >> 1;   // edge index for gather/scatter phases
    const int part   = lane & 1;    // which 24-feature half this lane owns

    // ---- hoist loop-invariant B fragments (W1) into registers ------------
    FragAB bfr[3][2];
    #pragma unroll
    for (int nt = 0; nt < 3; ++nt) {
        const unsigned short* rowB = &sW1[nt * 16 + nl][0];
        bfr[nt][0].q[0] = *(const uint4*)(rowB + kb);
        bfr[nt][0].q[1] = *(const uint4*)(rowB + kb + 16);
        bfr[nt][1].q[0] = *(const uint4*)(rowB + 32 + kb);
        bfr[nt][1].q[1] = *(const uint4*)(rowB + 32 + kb + 16);
    }
    // ---- hoist loop-invariant bias / W2 scalars --------------------------
    float bn[3], wn[3];
    #pragma unroll
    for (int nt = 0; nt < 3; ++nt) {
        bn[nt] = b1[nt * 16 + nl];
        wn[nt] = W2[nt * 16 + nl];
    }

    for (int tile = blockIdx.x * WAVES + wv; tile < nTiles;
         tile += gridDim.x * WAVES) {
        const long eBase = (long)tile * TILE_M;
        long eIdx = eBase + m_pair;
        const bool evalid = (eIdx < E);
        if (!evalid) eIdx = E - 1;

        const int sN = src[eIdx];
        const int dN = dst[eIdx];

        // -------- gather h_src / h_dst halves, compute h2, stage to LDS ----
        float hs[24];
        {
            const float* ps = h + (long)sN * DD + part * 24;
            const float* pd = h + (long)dN * DD + part * 24;
            float h2v[24];
            #pragma unroll
            for (int q = 0; q < 6; ++q) {
                float4 a = ((const float4*)ps)[q];
                float4 b = ((const float4*)pd)[q];
                hs[4*q+0] = a.x;  hs[4*q+1] = a.y;
                hs[4*q+2] = a.z;  hs[4*q+3] = a.w;
                h2v[4*q+0] = a.x * b.x;  h2v[4*q+1] = a.y * b.y;
                h2v[4*q+2] = a.z * b.z;  h2v[4*q+3] = a.w * b.w;
            }
            unsigned int* row = (unsigned int*)&sH2[wv][m_pair][part * 24];
            #pragma unroll
            for (int q = 0; q < 12; ++q)
                row[q] = pack_bf16(h2v[2*q], h2v[2*q+1]);
            // zero-pad k = 48..63 for the second (partial) k-step
            unsigned int* padp = (unsigned int*)&sH2[wv][m_pair][48 + part * 8];
            padp[0] = padp[1] = padp[2] = padp[3] = 0u;
        }
        asm volatile("s_wait_dscnt 0" ::: "memory");

        // -------- A fragments (16x32 bf16 layout: K-pairs are 16B blocks) --
        FragAB a0, a1;
        {
            const unsigned short* rowA = &sH2[wv][nl][0];
            a0.q[0] = *(const uint4*)(rowA + kb);
            a0.q[1] = *(const uint4*)(rowA + kb + 16);
            a1.q[0] = *(const uint4*)(rowA + 32 + kb);
            a1.q[1] = *(const uint4*)(rowA + 32 + kb + 16);
        }

        // -------- 3 N-tiles x 2 K-steps of WMMA (B already in registers) ---
        v8f acc[3];
        #pragma unroll
        for (int nt = 0; nt < 3; ++nt) {
            v8f c = {};
            c = __builtin_amdgcn_wmma_f32_16x16x32_bf16(
                    false, a0.v, false, bfr[nt][0].v, (short)0, c, false, false);
            c = __builtin_amdgcn_wmma_f32_16x16x32_bf16(
                    false, a1.v, false, bfr[nt][1].v, (short)0, c, false, false);
            acc[nt] = c;
        }

        // -------- epilogue: bias + PReLU + W2 dot -> per-lane partials -----
        #pragma unroll
        for (int r = 0; r < 8; ++r) {
            float p = 0.0f;
            #pragma unroll
            for (int nt = 0; nt < 3; ++nt) {
                float t = acc[nt][r] + bn[nt];
                t = (t >= 0.0f) ? t : alpha * t;
                p += t * wn[nt];
            }
            sP[wv][hi][r][nl] = p;      // LDS transpose for the reduction
        }
        asm volatile("s_wait_dscnt 0" ::: "memory");

        // -------- lanes 0..15: reduce 16 partials for edge m = lane --------
        if (lane < TILE_M) {
            const int m = lane;
            const float4* q4 = (const float4*)&sP[wv][m >> 3][m & 7][0];
            float4 s0 = q4[0], s1 = q4[1], s2 = q4[2], s3 = q4[3];
            float s = ((s0.x + s0.y) + (s0.z + s0.w))
                    + ((s1.x + s1.y) + (s1.z + s1.w))
                    + ((s2.x + s2.y) + (s2.z + s2.w))
                    + ((s3.x + s3.y) + (s3.z + s3.w));
            float g = tanhf(s + b2);
            sG[wv][m] = g;
            if (eBase + m < E) e_out[eBase + m] = g;   // coalesced e store
        }
        asm volatile("s_wait_dscnt 0" ::: "memory");
        const float ge = sG[wv][m_pair];

        // -------- scatter: z[dst] += h_src * g (HW f32 atomics, L2) --------
        if (evalid) {
            float* zp = z + (long)dN * DD + part * 24;
            #pragma unroll
            for (int j = 0; j < 24; ++j)
                __hip_atomic_fetch_add(&zp[j], hs[j] * ge,
                                       __ATOMIC_RELAXED,
                                       __HIP_MEMORY_SCOPE_AGENT);
        }
    }
}

extern "C" void kernel_launch(void* const* d_in, const int* in_sizes, int n_in,
                              void* d_out, int out_size, void* d_ws, size_t ws_size,
                              hipStream_t stream) {
    const float* h     = (const float*)d_in[0];
    const int*   src   = (const int*)d_in[1];
    const int*   dst   = (const int*)d_in[2];
    const float* W1    = (const float*)d_in[3];
    const float* b1    = (const float*)d_in[4];
    const float* alpha = (const float*)d_in[5];
    const float* W2    = (const float*)d_in[6];
    const float* b2    = (const float*)d_in[7];

    const long N = in_sizes[0] / DD;
    const long E = in_sizes[1];

    float* z     = (float*)d_out;
    float* e_out = z + N * DD;          // tuple output: z then e, flat

    // z is accumulated with atomics: clear it every launch
    zero_kernel<<<2048, 256, 0, stream>>>(z, N * DD);

    const int nTiles = (int)((E + TILE_M - 1) / TILE_M);
    int blocks = (nTiles + WAVES - 1) / WAVES;
    if (blocks > 4096) blocks = 4096;   // persistent grid-stride over tiles
    falayer_kernel<<<blocks, 256, 0, stream>>>(
        h, src, dst, W1, b1, alpha, W2, b2, z, e_out, E, nTiles);
}
